// OnlineBCELoss_6511170421617
// MI455X (gfx1250) — compile-verified, matrix operand order checked by reading, not run on hardware
//
#include <hip/hip_runtime.h>
#include <hip/hip_bf16.h>
#include <math.h>

// OnlineBCELoss fused pipeline for MI455X (gfx1250, wave32, WMMA):
//  K1 normalize: an/pn (f16) + cos_pos (f32)
//  K2 simmax:    dist = an @ pn^T via v_wmma_f32_16x16x32_f16; B tiles staged
//                through LDS with global_load_async_to_lds_b128 (double
//                buffered, ASYNCcnt-synced); fused row-max, diagonal mask
//                hoisted behind a scalar (readfirstlane) branch.
//  K3/K4:        BCEWithLogits mean, deterministic two-stage reduction.

typedef __attribute__((ext_vector_type(16))) _Float16 v16h;
typedef __attribute__((ext_vector_type(4)))  _Float16 v4h;
typedef __attribute__((ext_vector_type(8)))  float    v8f;

#define DIMK   128        // feature dim D
#define NCHUNK 8          // column chunks in the GEMM grid
#define EPSR   1e-8f
#define TILE_BYTES 4096   // 16 cols x 128 halfs x 2B

// ---------------------------------------------------------------- K1: normalize
__global__ __launch_bounds__(256) void normalize_kernel(
    const float* __restrict__ anchor, const float* __restrict__ positive,
    _Float16* __restrict__ an, _Float16* __restrict__ pn,
    float* __restrict__ cos_pos, int B)
{
    const int wave = threadIdx.x >> 5;
    const int lane = threadIdx.x & 31;
    const int row  = blockIdx.x * 8 + wave;
    if (row >= B) return;

    const float4 a = ((const float4*)(anchor   + (size_t)row * DIMK))[lane];
    const float4 p = ((const float4*)(positive + (size_t)row * DIMK))[lane];

    float na2 = a.x*a.x + a.y*a.y + a.z*a.z + a.w*a.w;
    float nb2 = p.x*p.x + p.y*p.y + p.z*p.z + p.w*p.w;
    float dab = a.x*p.x + a.y*p.y + a.z*p.z + a.w*p.w;

    #pragma unroll
    for (int off = 16; off >= 1; off >>= 1) {
        na2 += __shfl_xor(na2, off, 32);
        nb2 += __shfl_xor(nb2, off, 32);
        dab += __shfl_xor(dab, off, 32);
    }
    const float na = sqrtf(na2), nb = sqrtf(nb2);
    const float ia = 1.0f / na,  ib = 1.0f / nb;   // ref normalizes with raw norm

    v4h ah, ph;
    ah.x = (_Float16)(a.x * ia); ah.y = (_Float16)(a.y * ia);
    ah.z = (_Float16)(a.z * ia); ah.w = (_Float16)(a.w * ia);
    ph.x = (_Float16)(p.x * ib); ph.y = (_Float16)(p.y * ib);
    ph.z = (_Float16)(p.z * ib); ph.w = (_Float16)(p.w * ib);
    ((v4h*)(an + (size_t)row * DIMK))[lane] = ah;
    ((v4h*)(pn + (size_t)row * DIMK))[lane] = ph;

    if (lane == 0)
        cos_pos[row] = dab / (fmaxf(na, EPSR) * fmaxf(nb, EPSR));
}

// ---------------------------------------------------------- K2: WMMA GEMM + rowmax
// Async copy of 16 bytes into LDS per thread. lds_addr is derived from the
// __shared__ object itself (ptrtoint of its address -> wave-relative LDS byte
// offset in the low 32 bits), so the buffer escapes into this asm (which has a
// memory clobber) and alias analysis must keep the subsequent ds reads.
__device__ __forceinline__ void async_tile_load(const _Float16* gsrc, unsigned lds_addr)
{
    asm volatile("global_load_async_to_lds_b128 %0, %1, off"
                 :: "v"(lds_addr), "v"(gsrc) : "memory");
}

// wait for ASYNCcnt==0 (plain memory clobber; no "m" operand — the AMDGPU
// backend cannot select inline-asm memory operands)
__device__ __forceinline__ void wait_async_zero()
{
    asm volatile("s_wait_asynccnt 0x0" ::: "memory");
}

// single v_max_num_f32 (skip fmaxf's canonicalize max(x,x); WMMA output is
// already canonical per ISA 7.12: RNE, denorms preserved)
__device__ __forceinline__ float fmax_raw(float a, float b)
{
    float r;
    asm("v_max_num_f32 %0, %1, %2" : "=v"(r) : "v"(a), "v"(b));
    return r;
}

__global__ __launch_bounds__(256) void simmax_kernel(
    const _Float16* __restrict__ an, const _Float16* __restrict__ pn,
    float* __restrict__ pmax, int B)
{
    __shared__ _Float16 bbuf[2][16 * DIMK];   // two 4 KB B-tile buffers

    const int lane    = threadIdx.x & 31;
    const int wave    = threadIdx.x >> 5;
    const int rowBase = blockIdx.x * 128 + wave * 16;
    const int cols    = B / NCHUNK;
    const int ntiles  = cols / 16;
    const int chunk0  = blockIdx.y * cols;
    const int m       = lane & 15;
    const int half    = lane >> 4;
    const int rowOff  = half << 3;           // +8 for upper half-wave
    // wave-uniform scalar copy for the diagonal-tile test (s_cmp, no EXEC div)
    const int rowBaseU = __builtin_amdgcn_readfirstlane(rowBase);

    // per-thread slice of the staged tile: row r (0..15), 16B chunk ck (0..15)
    const int r  = threadIdx.x >> 4;
    const int ck = threadIdx.x & 15;
    const _Float16* gtile0 = pn + (size_t)(chunk0 + r) * DIMK + ck * 8;
    // LDS byte address of this thread's 16B slot, derived from bbuf (escapes)
    const unsigned ldsSlot =
        (unsigned)(uintptr_t)(&bbuf[0][0]) + (unsigned)(r * DIMK + ck * 8) * 2u;

    // resident A fragments for this wave's 16-row strip (K = 0..127)
    const _Float16* aRow = an + (size_t)(rowBase + m) * DIMK + half * 16;
    const v16h a0 = *(const v16h*)(aRow +  0);
    const v16h a1 = *(const v16h*)(aRow + 32);
    const v16h a2 = *(const v16h*)(aRow + 64);
    const v16h a3 = *(const v16h*)(aRow + 96);

    float cmax[8];
    #pragma unroll
    for (int v = 0; v < 8; ++v) cmax[v] = -INFINITY;

    // prologue: stage tile 0 into buffer 0
    async_tile_load(gtile0, ldsSlot);

    #pragma unroll 2
    for (int t = 0; t < ntiles; ++t) {
        const int cur = t & 1;

        wait_async_zero();            // my async writes to LDS landed
        __syncthreads();              // everyone's landed; old reads drained

        if (t + 1 < ntiles)           // stage next tile into the other buffer
            async_tile_load(gtile0 + (size_t)(t + 1) * 16 * DIMK,
                            ldsSlot + (unsigned)((1 - cur) * TILE_BYTES));

        // B fragments from LDS: lane -> col (m), 16 consecutive K at half*16
        const _Float16* bp = &bbuf[cur][m * DIMK + half * 16];
        const v16h b0 = *(const v16h*)(bp +  0);
        const v16h b1 = *(const v16h*)(bp + 32);
        const v16h b2 = *(const v16h*)(bp + 64);
        const v16h b3 = *(const v16h*)(bp + 96);

        v8f c = {0.f, 0.f, 0.f, 0.f, 0.f, 0.f, 0.f, 0.f};
        c = __builtin_amdgcn_wmma_f32_16x16x32_f16(false, a0, false, b0, (short)0, c, false, false);
        c = __builtin_amdgcn_wmma_f32_16x16x32_f16(false, a1, false, b1, (short)0, c, false, false);
        c = __builtin_amdgcn_wmma_f32_16x16x32_f16(false, a2, false, b2, (short)0, c, false, false);
        c = __builtin_amdgcn_wmma_f32_16x16x32_f16(false, a3, false, b3, (short)0, c, false, false);

        const int colBase = chunk0 + t * 16;
        if (colBase == rowBaseU) {
            // the single tile containing diagonal elements: mask them to 0
            // (dist * (1-eye)); element v at lane L is (row v+rowOff, col m)
            #pragma unroll
            for (int v = 0; v < 8; ++v)
                cmax[v] = fmax_raw(cmax[v], (m == v + rowOff) ? 0.0f : c[v]);
        } else {
            #pragma unroll
            for (int v = 0; v < 8; ++v)
                cmax[v] = fmax_raw(cmax[v], c[v]);
        }
    }

    // group-of-16 max reduction (xor masks < 16 stay within each half-wave)
    #pragma unroll
    for (int v = 0; v < 8; ++v) {
        #pragma unroll
        for (int off = 1; off < 16; off <<= 1)
            cmax[v] = fmax_raw(cmax[v], __shfl_xor(cmax[v], off, 32));
    }
    // writer lanes: lane m==v (lower half -> row v, upper half -> row v+8)
    #pragma unroll
    for (int v = 0; v < 8; ++v) {
        if (m == v) {
            const int row = rowBase + v + rowOff;
            pmax[(size_t)blockIdx.y * B + row] = cmax[v];
        }
    }
}

// ---------------------------------------------------------------- K3/K4: loss
__device__ __forceinline__ float softplus_f(float x) {
    // logaddexp(0, x) = max(x,0) + log1p(exp(-|x|))
    return fmaxf(x, 0.0f) + log1pf(expf(-fabsf(x)));
}

__global__ __launch_bounds__(256) void loss_partial_kernel(
    const float* __restrict__ cos_pos, const float* __restrict__ pmax,
    float* __restrict__ bsum, int B)
{
    const int idx = blockIdx.x * 256 + threadIdx.x;
    float l = 0.0f;
    if (idx < B) {
        const float cp = cos_pos[idx];
        float cn = -INFINITY;
        #pragma unroll
        for (int c = 0; c < NCHUNK; ++c)
            cn = fmaxf(cn, pmax[(size_t)c * B + idx]);
        // pos term: softplus(x) - x ; neg term: softplus(x)
        l = (softplus_f(cp) - cp) + softplus_f(cn);
    }
    __shared__ float sdata[256];
    sdata[threadIdx.x] = l;
    __syncthreads();
    #pragma unroll
    for (int s = 128; s > 0; s >>= 1) {
        if (threadIdx.x < s) sdata[threadIdx.x] += sdata[threadIdx.x + s];
        __syncthreads();
    }
    if (threadIdx.x == 0) bsum[blockIdx.x] = sdata[0];
}

__global__ void loss_final_kernel(const float* __restrict__ bsum,
                                  float* __restrict__ out, int nblocks, float invN)
{
    if (threadIdx.x == 0) {
        float s = 0.0f;
        for (int i = 0; i < nblocks; ++i) s += bsum[i];  // fixed order: deterministic
        out[0] = s * invN;
    }
}

// ---------------------------------------------------------------- launch
extern "C" void kernel_launch(void* const* d_in, const int* in_sizes, int n_in,
                              void* d_out, int out_size, void* d_ws, size_t ws_size,
                              hipStream_t stream) {
    const float* anchor   = (const float*)d_in[0];
    const float* positive = (const float*)d_in[1];
    const int B = in_sizes[0] / DIMK;   // 8192

    // workspace layout
    char* ws = (char*)d_ws;
    _Float16* an      = (_Float16*)ws;                       // B*D halfs (2 MB)
    _Float16* pn      = an + (size_t)B * DIMK;               // B*D halfs (2 MB)
    float*    cos_pos = (float*)(pn + (size_t)B * DIMK);     // B floats
    float*    pmax    = cos_pos + B;                         // NCHUNK*B floats
    float*    bsum    = pmax + (size_t)NCHUNK * B;           // B/256 floats
    float*    out     = (float*)d_out;

    normalize_kernel<<<B / 8, 256, 0, stream>>>(anchor, positive, an, pn, cos_pos, B);

    dim3 grid(B / 128, NCHUNK);
    simmax_kernel<<<grid, 256, 0, stream>>>(an, pn, pmax, B);

    const int nb = (B + 255) / 256;
    loss_partial_kernel<<<nb, 256, 0, stream>>>(cos_pos, pmax, bsum, B);
    loss_final_kernel<<<1, 32, 0, stream>>>(bsum, out, nb, 1.0f / (2.0f * (float)B));
}
